// MultiHeadGeneralizedPooling_35545149342225
// MI455X (gfx1250) — matrix operand order repaired and checked by reading, not compiled, not run on hardware
//
#include <hip/hip_runtime.h>
#include <cstdint>
#include <cstddef>

// ---------------------------------------------------------------------------
// MultiHeadGeneralizedPooling for MI455X (gfx1250, wave32, WMMA bf16)
//   B=128 S=512 D=768 NH=8 HD=96 HID=384, ROWS = B*S = 65536
// - Projection GEMM (77 GFLOP, compute-bound): v_wmma_f32_16x16x32_bf16,
//   2x2 register tiling per wave (4 WMMAs per 4 fragment loads).
// - Per-head MLP: fused relu GEMM chain, T1 restaged through LDS.
// - Pooling: single-pass online masked softmax + weighted sum, S split
//   across 4 sub-groups merged via LDS (one read of As instead of two).
// Xb (bf16, 100MB) fits in the 192MB L2 -> A-matrix re-reads across column
// tiles are L2 hits; HBM traffic stays near one pass per tensor.
// ---------------------------------------------------------------------------

#define B_    128
#define S_    512
#define D_    768
#define NH_   8
#define HD_   96
#define HID_  384
#define ROWS_ (B_ * S_)   // 65536

typedef __attribute__((ext_vector_type(16))) __bf16 v16bf;
typedef __attribute__((ext_vector_type(8)))  float  v8f;

// ---- WMMA fragment loaders (contiguous k = 16*(L/16)+s in A and B, wave32) -------------
// A (16x32 bf16, row-major source): lane L: M = L%16, K-base = (L<16?0:16)
//   vgpr j 0..7 -> K = base+2j
__device__ __forceinline__ v16bf load_frag_a(const __bf16* __restrict__ A,
                                             int lda, int row0, int k0, int lane) {
  int m  = lane & 15;
  int kb = (lane >> 4) << 4;          // 0 or 16
  const __bf16* p = A + (size_t)(row0 + m) * (size_t)lda + (size_t)(k0 + kb);
  v16bf a;
#pragma unroll
  for (int j = 0; j < 4; ++j) { a[2 * j]     = p[2 * j];      a[2 * j + 1] = p[2 * j + 1]; }
#pragma unroll
  for (int j = 0; j < 4; ++j) { a[8 + 2 * j] = p[8 + 2 * j]; a[9 + 2 * j] = p[8 + 2 * j + 1]; }
  return a;
}

// B (32x16 bf16) from weights stored transposed as Bw[n][k] (row-major):
//   lane L: N = L%16 ; K-base = (L<16 ? 0 : 16) ; vgpr j -> K = base+2j
__device__ __forceinline__ v16bf load_frag_bT(const __bf16* __restrict__ Bw,
                                              int ldb, int col0, int k0, int lane) {
  int n  = lane & 15;
  int kb = (lane >> 4) << 4;          // 0 or 16
  const __bf16* p = Bw + (size_t)(col0 + n) * (size_t)ldb + (size_t)(k0 + kb);
  v16bf b;
#pragma unroll
  for (int j = 0; j < 8; ++j) { b[2 * j] = p[2 * j]; b[2 * j + 1] = p[2 * j + 1]; }
  return b;
}

__device__ __forceinline__ v8f wmma_bf16(v16bf a, v16bf b, v8f c) {
  return __builtin_amdgcn_wmma_f32_16x16x32_bf16(false, a, false, b,
                                                 (short)0, c, false, false);
}

// ---- Kernel 0: f32 -> bf16 conversion -------------------------------------
__global__ void cvt_bf16_kernel(const float* __restrict__ src,
                                __bf16* __restrict__ dst, size_t n) {
  size_t i      = (size_t)blockIdx.x * blockDim.x + threadIdx.x;
  size_t stride = (size_t)gridDim.x * blockDim.x;
  for (; i < n; i += stride) dst[i] = (__bf16)src[i];
}

// ---- Kernel 1: Hi = X @ P^T + bP  (65536x768 * 768x768) -------------------
// 8 waves / block arranged 4(row) x 2(col); each wave owns a 32x32 output
// tile (2x2 WMMA tiles -> 4 WMMAs per 4 fragment loads per k-step).
// Block tile = 128 rows x 64 cols. Writes Hi (f32) and Hib (bf16),
// layout [head][row][h].
__device__ __forceinline__ void store_tile_proj(
    v8f c, int row0, int col0, int lane,
    const float* __restrict__ bP,
    float* __restrict__ Hi, __bf16* __restrict__ Hib) {
  int n   = lane & 15;
  int mb  = (lane >> 4) << 3;
  int col = col0 + n;                 // 96 = 6*16: a 16-col tile is one head
  int head = col / HD_, h = col % HD_;
  float bias = bP[col];
  size_t base = (size_t)head * ROWS_ * HD_ + (size_t)h;
#pragma unroll
  for (int r = 0; r < 8; ++r) {
    int row  = row0 + mb + r;
    float v  = c[r] + bias;
    Hi [base + (size_t)row * HD_] = v;
    Hib[base + (size_t)row * HD_] = (__bf16)v;
  }
}

__global__ __launch_bounds__(256) void proj_gemm_kernel(
    const __bf16* __restrict__ Xb, const __bf16* __restrict__ Pb,
    const float* __restrict__ bP,
    float* __restrict__ Hi, __bf16* __restrict__ Hib) {
  int lane = threadIdx.x & 31;
  int wave = threadIdx.x >> 5;        // 0..7
  int wr   = wave >> 1;               // 0..3 row group
  int wc   = wave & 1;                // 0..1 col group
  int row0 = blockIdx.x * 128 + wr * 32;
  int col0 = blockIdx.y * 64  + wc * 32;

  v8f c00 = {}, c01 = {}, c10 = {}, c11 = {};
#pragma unroll 2
  for (int k0 = 0; k0 < D_; k0 += 32) {
    v16bf a0 = load_frag_a (Xb, D_, row0,      k0, lane);
    v16bf a1 = load_frag_a (Xb, D_, row0 + 16, k0, lane);
    v16bf b0 = load_frag_bT(Pb, D_, col0,      k0, lane);
    v16bf b1 = load_frag_bT(Pb, D_, col0 + 16, k0, lane);
    c00 = wmma_bf16(a0, b0, c00);
    c01 = wmma_bf16(a0, b1, c01);
    c10 = wmma_bf16(a1, b0, c10);
    c11 = wmma_bf16(a1, b1, c11);
  }
  store_tile_proj(c00, row0,      col0,      lane, bP, Hi, Hib);
  store_tile_proj(c01, row0,      col0 + 16, lane, bP, Hi, Hib);
  store_tile_proj(c10, row0 + 16, col0,      lane, bP, Hi, Hib);
  store_tile_proj(c11, row0 + 16, col0 + 16, lane, bP, Hi, Hib);
}

// ---- Kernel 2: A = relu(Hi @ W1^T + b1) @ W2^T + b2  (per head) -----------
// 4 waves / block, each wave owns one 16-row tile of one head.
// T1 (16x384 bf16) restaged per-wave through LDS (C-frag -> A-frag relayout).
__global__ __launch_bounds__(128) void mlp_kernel(
    const __bf16* __restrict__ Hib, const __bf16* __restrict__ W1b,
    const __bf16* __restrict__ W2b, const float* __restrict__ b1,
    const float* __restrict__ b2, float* __restrict__ As) {
  __shared__ __bf16 t1s[4 * 16 * HID_];          // 48 KB

  int lane = threadIdx.x & 31;
  int wave = threadIdx.x >> 5;
  int head = blockIdx.y;
  int row0 = (blockIdx.x * 4 + wave) * 16;

  const __bf16* Ah  = Hib + (size_t)head * ROWS_ * HD_;
  const __bf16* W1h = W1b + (size_t)head * HID_ * HD_;   // [f][k] row-major
  const __bf16* W2h = W2b + (size_t)head * HD_ * HID_;   // [h][f] row-major
  __bf16* t1 = t1s + wave * 16 * HID_;

  int n  = lane & 15;
  int mb = (lane >> 4) << 3;

  // Stage 1: T1 = relu(Hi_tile @ W1^T + b1)  -> LDS (bf16)
  v16bf afr[3];
#pragma unroll
  for (int kk = 0; kk < 3; ++kk) afr[kk] = load_frag_a(Ah, HD_, row0, kk * 32, lane);

  for (int ct = 0; ct < HID_; ct += 16) {
    v8f c = {};
#pragma unroll
    for (int kk = 0; kk < 3; ++kk) {
      v16bf b = load_frag_bT(W1h, HD_, ct, kk * 32, lane);
      c = wmma_bf16(afr[kk], b, c);
    }
    float bias = b1[head * HID_ + ct + n];
#pragma unroll
    for (int r = 0; r < 8; ++r) {
      float v = c[r] + bias;
      v = v > 0.f ? v : 0.f;                              // relu
      t1[(size_t)(mb + r) * HID_ + (size_t)(ct + n)] = (__bf16)v;
    }
  }

  // Stage 2: A = T1 @ W2^T + b2   (K = 384, N = 96)
  size_t obase = ((size_t)head * ROWS_ + (size_t)row0) * HD_;
  for (int ct = 0; ct < HD_; ct += 16) {
    v8f c = {};
#pragma unroll 4
    for (int k0 = 0; k0 < HID_; k0 += 32) {
      v16bf a = load_frag_a (t1,  HID_, 0,  k0, lane);    // ds_load from LDS
      v16bf b = load_frag_bT(W2h, HID_, ct, k0, lane);
      c = wmma_bf16(a, b, c);
    }
    float bias = b2[head * HD_ + ct + n];
#pragma unroll
    for (int r = 0; r < 8; ++r)
      As[obase + (size_t)(mb + r) * HD_ + (size_t)(ct + n)] = c[r] + bias;
  }
}

// ---- Kernel 3: single-pass masked online softmax + weighted sum -----------
// One block per (batch, head). 512 threads = 4 sub-groups of 128; sub-group g
// handles tokens s in [g*128, (g+1)*128); lanes 0..95 of each sub-group own
// one HD channel. Partial (m, denom, num) triples merged through LDS.
__global__ __launch_bounds__(512) void pool_kernel(
    const float* __restrict__ As, const float* __restrict__ Hi,
    const float* __restrict__ mask, float* __restrict__ out) {
  __shared__ float sm[4][HD_];
  __shared__ float sd[4][HD_];
  __shared__ float sn[4][HD_];

  int b    = blockIdx.x, head = blockIdx.y;
  int g    = threadIdx.x >> 7;        // sub-group 0..3
  int h    = threadIdx.x & 127;       // channel (0..95 active)
  bool act = (h < HD_);

  size_t base = ((size_t)head * ROWS_ + (size_t)b * S_) * HD_ + (size_t)h;
  const float* mrow = mask + (size_t)b * S_;

  float m = -3.402823e38f, denom = 0.f, num = 0.f;
  if (act) {
    int s0 = g * (S_ / 4), s1 = s0 + (S_ / 4);
    for (int s = s0; s < s1; ++s) {
      if (mrow[s] > 0.f) {
        float a  = As[base + (size_t)s * HD_];
        float hi = Hi[base + (size_t)s * HD_];
        float mn = fmaxf(m, a);
        float cr = __expf(m - mn);    // first hit: exp(-inf) = 0
        float e  = __expf(a - mn);
        denom = denom * cr + e;
        num   = num   * cr + e * hi;
        m = mn;
      }
    }
    sm[g][h] = m; sd[g][h] = denom; sn[g][h] = num;
  }
  __syncthreads();

  if (g == 0 && act) {
    float M = fmaxf(fmaxf(sm[0][h], sm[1][h]), fmaxf(sm[2][h], sm[3][h]));
    float D = 0.f, N = 0.f;
#pragma unroll
    for (int q = 0; q < 4; ++q) {
      float dq = sd[q][h];
      if (dq > 0.f) {                 // skip fully-masked chunks (m = -inf)
        float cr = __expf(sm[q][h] - M);
        D += dq * cr;
        N += sn[q][h] * cr;
      }
    }
    out[(size_t)b * (NH_ * HD_) + (size_t)head * HD_ + (size_t)h] =
        D > 0.f ? N / D : 0.f;
  }
}

// ---------------------------------------------------------------------------
extern "C" void kernel_launch(void* const* d_in, const int* in_sizes, int n_in,
                              void* d_out, int out_size, void* d_ws, size_t ws_size,
                              hipStream_t stream) {
  const float* X    = (const float*)d_in[0];  // (B,S,D)
  const float* mask = (const float*)d_in[1];  // (B,S)
  const float* P    = (const float*)d_in[2];  // (NH,HD,D) == (768,768) flat
  const float* bP   = (const float*)d_in[3];  // (NH,HD)
  const float* W1   = (const float*)d_in[4];  // (NH,HID,HD)
  const float* b1   = (const float*)d_in[5];  // (NH,HID)
  const float* W2   = (const float*)d_in[6];  // (NH,HD,HID)
  const float* b2   = (const float*)d_in[7];  // (NH,HD)
  float* out = (float*)d_out;                 // (B, NH*HD)

  // Workspace layout (all offsets 256B-aligned by construction)
  char* ws = (char*)d_ws;
  size_t offXb  = 0;                                              // 100,663,296
  size_t offPb  = offXb  + (size_t)ROWS_ * D_ * 2;                // +1,179,648
  size_t offW1  = offPb  + (size_t)D_ * D_ * 2;                   // +589,824
  size_t offW2  = offW1  + (size_t)NH_ * HID_ * HD_ * 2;          // +589,824
  size_t offHi  = offW2  + (size_t)NH_ * HD_ * HID_ * 2;          // f32 201 MB
  size_t offHib = offHi  + (size_t)NH_ * ROWS_ * HD_ * 4;         // bf16 100 MB
  size_t offAs  = offHib + (size_t)NH_ * ROWS_ * HD_ * 2;         // f32 201 MB

  __bf16* Xb  = (__bf16*)(ws + offXb);
  __bf16* Pb  = (__bf16*)(ws + offPb);
  __bf16* W1b = (__bf16*)(ws + offW1);
  __bf16* W2b = (__bf16*)(ws + offW2);
  float*  Hi  = (float*) (ws + offHi);
  __bf16* Hib = (__bf16*)(ws + offHib);
  float*  As  = (float*) (ws + offAs);

  // 0) bf16 conversions
  cvt_bf16_kernel<<<2048, 256, 0, stream>>>(X,  Xb,  (size_t)ROWS_ * D_);
  cvt_bf16_kernel<<<512,  256, 0, stream>>>(P,  Pb,  (size_t)D_ * D_);
  cvt_bf16_kernel<<<256,  256, 0, stream>>>(W1, W1b, (size_t)NH_ * HID_ * HD_);
  cvt_bf16_kernel<<<256,  256, 0, stream>>>(W2, W2b, (size_t)NH_ * HD_ * HID_);

  // 1) Hi = X @ P^T + bP     grid: 512 row-blocks x 12 col-blocks (128x64)
  proj_gemm_kernel<<<dim3(ROWS_ / 128, D_ / 64), 256, 0, stream>>>(Xb, Pb, bP, Hi, Hib);

  // 2) A = relu(Hi@W1^T+b1)@W2^T + b2   grid: 1024 row-blocks x 8 heads
  mlp_kernel<<<dim3(ROWS_ / 64, NH_), 128, 0, stream>>>(Hib, W1b, W2b, b1, b2, As);

  // 3) single-pass masked softmax over S + weighted sum -> pooled (B, 768)
  pool_kernel<<<dim3(B_, NH_), 512, 0, stream>>>(As, Hi, mask, out);
}
